// Net_17532056502451
// MI455X (gfx1250) — matrix operand order repaired and checked by reading, not compiled, not run on hardware
//
#include <hip/hip_runtime.h>
#include <hip/hip_bf16.h>

#define IDIM 80
#define ODIM 80
#define HDIM 512
#define NSEL 159      // 2*ODIM-1 selector shifts
#define NENC 81       // encoder shifts
#define KPAD 96       // K padded to 3 x 32 for WMMA f16
#define W8W  192      // padded row width of shifted W_enc copies
#define ESTR 84       // energy buffer row stride (floats)
#define TEMPER 0.7f

typedef _Float16 h16v __attribute__((ext_vector_type(16)));
typedef _Float16 h8v  __attribute__((ext_vector_type(8)));
typedef float    f8v  __attribute__((ext_vector_type(8)));

// ---------------------------------------------------------------------------
// K_init: copy residuals, build 8 byte-shifted f16 copies of W_enc so every
// WMMA B-fragment load is a 16B-aligned global_load_b128.
//   W8[r][row][cp] = (0 <= cp-r < 160) ? f16(W_enc[row][cp-r]) : 0
// ---------------------------------------------------------------------------
__global__ void k_init(const float* __restrict__ x, const float* __restrict__ y,
                       const float* __restrict__ W_enc,
                       float* __restrict__ x_res, float* __restrict__ y_res,
                       _Float16* __restrict__ W8, int P) {
    int idx = blockIdx.x * 256 + threadIdx.x;
    const int n8 = 8 * HDIM * W8W;
    if (idx < n8) {
        int r   = idx / (HDIM * W8W);
        int rem = idx - r * (HDIM * W8W);
        int row = rem / W8W;
        int cp  = rem % W8W;
        int c   = cp - r;
        W8[idx] = (c >= 0 && c < 2 * IDIM) ? (_Float16)W_enc[row * (2 * IDIM) + c]
                                           : (_Float16)0.0f;
    }
    if (idx < P * ODIM) {
        x_res[idx] = x[idx];
        y_res[idx] = y[idx];
    }
}

// Deterministic keep-count (keep = y != 0), per-block integer partials.
__global__ void k_count(const float* __restrict__ y, int n, int* __restrict__ cntPart) {
    __shared__ int red[256];
    int tid = threadIdx.x;
    int idx = blockIdx.x * 256 + tid;
    red[tid] = (idx < n && y[idx] != 0.0f) ? 1 : 0;
    __syncthreads();
    for (int off = 128; off > 0; off >>= 1) {
        if (tid < off) red[tid] += red[tid + off];
        __syncthreads();
    }
    if (tid == 0) cntPart[blockIdx.x] = red[0];
}

// ---------------------------------------------------------------------------
// K1: selector (cosine sim over 159 shifts) + softmax attention + y_att,
// reverse-shift x_ele, x_res update.  One block per position.
// ---------------------------------------------------------------------------
__global__ void k_selector(float* __restrict__ x_res, const float* __restrict__ y_res,
                           float* __restrict__ y_att, int P) {
    __shared__ float xr[ODIM], yr[ODIM], sim[NSEL];
    __shared__ float ya[ODIM], att[ODIM], yatt[ODIM];
    __shared__ int   s_theta;
    __shared__ float s_red[2];

    const int p = blockIdx.x;
    const int t = threadIdx.x;
    if (p >= P) return;

    if (t < ODIM) { xr[t] = x_res[p * ODIM + t]; yr[t] = y_res[p * ODIM + t]; }
    __syncthreads();

    if (t < NSEL) {
        float ny2 = 0.0f;
        for (int c = 0; c < ODIM; ++c) ny2 += yr[c] * yr[c];
        float dot = 0.0f, nx2 = 0.0f;
        for (int c = 0; c < ODIM; ++c) {
            int ix = t + c - (ODIM - 1);
            float xv = (ix >= 0 && ix < ODIM) ? xr[ix] : 0.0f;
            dot += xv * yr[c];
            nx2 += xv * xv;
        }
        float den = sqrtf(nx2) * sqrtf(ny2);
        sim[t] = (den == 0.0f) ? 0.0f : dot / den;
    }
    __syncthreads();
    if (t == 0) {                      // first-max argmax (matches jnp.argmax)
        float best = sim[0]; int bi = 0;
        for (int s = 1; s < NSEL; ++s) if (sim[s] > best) { best = sim[s]; bi = s; }
        s_theta = bi;
    }
    __syncthreads();
    const int th = s_theta;
    if (t < ODIM) {
        int ix = th + t - (ODIM - 1);
        float v = (ix >= 0 && ix < ODIM) ? xr[ix] : 0.0f;
        ya[t]  = v;
        att[t] = v * yr[t] * (1.0f / TEMPER);
    }
    __syncthreads();
    if (t == 0) { float mx = att[0]; for (int c = 1; c < ODIM; ++c) mx = fmaxf(mx, att[c]); s_red[0] = mx; }
    __syncthreads();
    if (t < ODIM) att[t] = expf(att[t] - s_red[0]);
    __syncthreads();
    if (t == 0) { float sm = 0.0f; for (int c = 0; c < ODIM; ++c) sm += att[c]; s_red[1] = sm; }
    __syncthreads();
    if (t < ODIM) {
        float v = ya[t] * (att[t] / s_red[1]);
        yatt[t] = v;
        y_att[p * ODIM + t] = v;
    }
    __syncthreads();
    if (t < ODIM) {                    // x_ele[j] = y_att[79 - theta + j] (windowed)
        int ix = (ODIM - 1) - th + t;
        float xe = (ix >= 0 && ix < ODIM) ? yatt[ix] : 0.0f;
        x_res[p * ODIM + t] = xr[t] - xe;
    }
}

// ---------------------------------------------------------------------------
// K2a: encoder energy via WMMA.  Grid = (P/16, 8): blockIdx.x = position
// tile, blockIdx.y = g owns shifts s == g (mod 8) — 512 workgroups instead of
// 64 so the WMMA work fills the chip.  Per shift: 32 hid tiles x 3 WMMA
// (K padded to 96).  C is initialized with b_enc so D = h + b; energy
// accumulates D^2, reduces across the 16 N-lanes with shfl_xor, writes to a
// global energy buffer (disjoint s-columns per blockIdx.y -> no races).
// ---------------------------------------------------------------------------
__global__ __launch_bounds__(256) void k_energy(
        const float* __restrict__ y_att, const float* __restrict__ b_enc,
        const _Float16* __restrict__ W8, float* __restrict__ energyBuf, int P) {
    __shared__ __align__(16) _Float16 s_ya16[16 * KPAD];
    __shared__ float s_benc[HDIM];

    const int tid = threadIdx.x;
    const int p0  = blockIdx.x * 16;
    const int g   = blockIdx.y;                   // shift residue class

    // Phase 0: stage y_att tile (f16, K-padded) and b_enc.
    for (int i = tid; i < 16 * KPAD; i += 256) {
        int m = i / KPAD, c = i % KPAD;
        float v = 0.0f;
        if (c < ODIM && (p0 + m) < P) v = y_att[(size_t)(p0 + m) * ODIM + c];
        s_ya16[i] = (_Float16)v;
    }
    for (int i = tid; i < HDIM; i += 256) s_benc[i] = b_enc[i];
    __syncthreads();

    const int lane  = tid & 31;
    const int wave  = tid >> 5;
    const int nIdx  = lane & 15;      // N (hid within tile) / M (A row)
    const int hiSel = lane >> 4;

    // A fragments (shared by every shift & hid tile).
    h16v afr[3];
    {
        const _Float16* ar = s_ya16 + (size_t)nIdx * KPAD;
        #pragma unroll
        for (int t = 0; t < 3; ++t) {
            const h8v* lo = (const h8v*)(ar + 32 * t + 8 * hiSel);
            const h8v* hi = (const h8v*)(ar + 32 * t + 16 + 8 * hiSel);
            h8v l = *lo, h = *hi;
            #pragma unroll
            for (int i = 0; i < 8; ++i) { afr[t][i] = l[i]; afr[t][i + 8] = h[i]; }
        }
    }

    const int cnt = (NENC - 1 - g) / 8 + 1;       // #shifts in this residue class
    for (int si = wave; si < cnt; si += 8) {
        const int s = g + 8 * si;
        const int r = s & 7;                      // alignment-fixing copy (== g&7)
        const _Float16* Wp = W8 + (size_t)r * HDIM * W8W + (IDIM - s + r);
        float acc[8] = {0, 0, 0, 0, 0, 0, 0, 0};
        for (int ht = 0; ht < HDIM / 16; ++ht) {
            const int row = ht * 16 + nIdx;
            const _Float16* bb = Wp + (size_t)row * W8W + 8 * hiSel;
            const float bv = s_benc[row];
            f8v c = {bv, bv, bv, bv, bv, bv, bv, bv};
            #pragma unroll
            for (int t = 0; t < 3; ++t) {
                const h8v* bp = (const h8v*)(bb + 32 * t);
                h8v bl = bp[0], bh = bp[2];
                h16v b;
                #pragma unroll
                for (int i = 0; i < 8; ++i) { b[i] = bl[i]; b[i + 8] = bh[i]; }
                c = __builtin_amdgcn_wmma_f32_16x16x32_f16(
                        false, afr[t], false, b, (short)0, c, false, false);
            }
            #pragma unroll
            for (int i = 0; i < 8; ++i) acc[i] += c[i] * c[i];
        }
        #pragma unroll
        for (int i = 0; i < 8; ++i) {             // reduce over 16 N-lanes
            float v = acc[i];
            v += __shfl_xor(v, 1, 16);
            v += __shfl_xor(v, 2, 16);
            v += __shfl_xor(v, 4, 16);
            v += __shfl_xor(v, 8, 16);
            acc[i] = v;
        }
        if (nIdx == 0) {
            const int mb = hiSel * 8;
            #pragma unroll
            for (int i = 0; i < 8; ++i) {
                int p = p0 + mb + i;
                if (p < P) energyBuf[(size_t)p * ESTR + s] = acc[i];
            }
        }
    }
}

// ---------------------------------------------------------------------------
// K2b: per-position first-max argmax over shifts + exact f32 recompute of
// h_sel for the winning shift.
// ---------------------------------------------------------------------------
__global__ __launch_bounds__(256) void k_argmax_hsel(
        const float* __restrict__ y_att, const float* __restrict__ W_enc,
        const float* __restrict__ b_enc, const float* __restrict__ energyBuf,
        float* __restrict__ h_sel, int* __restrict__ h_ind, int P) {
    __shared__ float s_ya32[16 * ODIM];
    __shared__ float s_benc[HDIM];
    __shared__ int   s_hind[16];

    const int tid = threadIdx.x;
    const int p0  = blockIdx.x * 16;

    for (int i = tid; i < 16 * ODIM; i += 256) {
        int m = i / ODIM, c = i % ODIM;
        s_ya32[i] = (p0 + m < P) ? y_att[(size_t)(p0 + m) * ODIM + c] : 0.0f;
    }
    for (int i = tid; i < HDIM; i += 256) s_benc[i] = b_enc[i];
    if (tid < 16) {
        int p = p0 + tid;
        int bi = 0;
        if (p < P) {
            const float* er = energyBuf + (size_t)p * ESTR;
            float best = er[0];
            for (int s = 1; s < NENC; ++s) {
                float e = er[s];
                if (e > best) { best = e; bi = s; }
            }
            h_ind[p] = bi;
        }
        s_hind[tid] = bi;
    }
    __syncthreads();

    for (int i = tid; i < 16 * HDIM; i += 256) {
        int m = i >> 9, hid = i & (HDIM - 1);
        if (p0 + m >= P) continue;
        const int ss = s_hind[m];
        const float* wrow = W_enc + (size_t)hid * (2 * IDIM) + (IDIM - ss);
        const float* yam  = s_ya32 + m * ODIM;
        float sum = s_benc[hid];
        for (int j = 0; j < ODIM; ++j) sum += wrow[j] * yam[j];
        h_sel[(size_t)(p0 + m) * HDIM + hid] = sum;
    }
}

// ---------------------------------------------------------------------------
// K3a: decoder x_ext = W_src @ h_sel + b_src.  Grid = (P/16, 5): blockIdx.y
// owns a 32-wide output tile -> 320 workgroups.
// ---------------------------------------------------------------------------
__global__ __launch_bounds__(256) void k_dec_gemm(
        const float* __restrict__ h_sel, const float* __restrict__ W_src,
        const float* __restrict__ b_src, float* __restrict__ x_ext, int P) {
    __shared__ float s_h[16 * HDIM];
    const int tid = threadIdx.x;
    const int p0  = blockIdx.x * 16;
    const int ob  = blockIdx.y * 32;

    for (int i = tid; i < 16 * HDIM; i += 256) {
        int m = i >> 9, k = i & (HDIM - 1);
        s_h[i] = (p0 + m < P) ? h_sel[(size_t)(p0 + m) * HDIM + k] : 0.0f;
    }
    __syncthreads();

    for (int i = tid; i < 16 * 32; i += 256) {
        int m = i >> 5, o = ob + (i & 31);
        const float* wr = W_src + (size_t)o * HDIM;
        const float* hm = s_h + m * HDIM;
        float sum = b_src[o];
        for (int k = 0; k < HDIM; ++k) sum += wr[k] * hm[k];
        if (p0 + m < P) x_ext[(size_t)(p0 + m) * (2 * ODIM) + o] = sum;
    }
}

// ---------------------------------------------------------------------------
// K3b: gather y_ele window, masked squared-error partial per block
// (deterministic), y_res update.
// ---------------------------------------------------------------------------
__global__ __launch_bounds__(256) void k_loss(
        const float* __restrict__ x_ext, const int* __restrict__ h_ind,
        const float* __restrict__ y_in, float* __restrict__ y_res,
        float* __restrict__ lossPart, int it, int P, int nblkLoss) {
    __shared__ float s_red[256];
    const int tid = threadIdx.x;
    const int idx = blockIdx.x * 256 + tid;

    float part = 0.0f;
    if (idx < P * ODIM) {
        int p = idx / ODIM, j = idx % ODIM;
        const int hv = h_ind[p];
        float ye   = x_ext[(size_t)p * (2 * ODIM) + (IDIM - hv) + j];
        float yr   = y_res[idx];
        float keep = (y_in[idx] != 0.0f) ? 1.0f : 0.0f;
        float d = ye - yr;
        part = d * d * keep;
        y_res[idx] = yr - ye;
    }
    s_red[tid] = part;
    __syncthreads();
    for (int off = 128; off > 0; off >>= 1) {
        if (tid < off) s_red[tid] += s_red[tid + off];
        __syncthreads();
    }
    if (tid == 0) lossPart[it * nblkLoss + blockIdx.x] = s_red[0];
}

__global__ void k_final(const float* __restrict__ lossPart, const int* __restrict__ cntPart,
                        int nblkLoss, int ncb, float* __restrict__ out) {
    if (threadIdx.x == 0 && blockIdx.x == 0) {
        int cnt = 0;
        for (int i = 0; i < ncb; ++i) cnt += cntPart[i];
        float mc = (float)(cnt > 1 ? cnt : 1);
        float acc = 0.0f;
        for (int it = 0; it < 5; ++it) {
            float s = 0.0f;
            for (int b = 0; b < nblkLoss; ++b) s += lossPart[it * nblkLoss + b];
            acc += s / mc;
        }
        out[0] = acc / 5.0f;
    }
}

// ---------------------------------------------------------------------------
extern "C" void kernel_launch(void* const* d_in, const int* in_sizes, int n_in,
                              void* d_out, int out_size, void* d_ws, size_t ws_size,
                              hipStream_t stream) {
    (void)n_in; (void)out_size; (void)ws_size;
    const float* x     = (const float*)d_in[0];
    const float* y     = (const float*)d_in[1];
    const float* W_enc = (const float*)d_in[2];
    const float* b_enc = (const float*)d_in[3];
    const float* W_src = (const float*)d_in[4];
    const float* b_src = (const float*)d_in[5];

    const int P        = in_sizes[0] / IDIM;        // B*T = 1024
    const int nblk16   = (P + 15) / 16;
    const int nblkLoss = (P * ODIM + 255) / 256;
    const int ncb      = nblkLoss;

    size_t off = 0;
    auto carve = [&](size_t bytes) -> void* {
        void* p = (char*)d_ws + off;
        off += (bytes + 255) & ~(size_t)255;
        return p;
    };
    float*    x_res    = (float*)carve((size_t)P * ODIM * 4);
    float*    y_res    = (float*)carve((size_t)P * ODIM * 4);
    float*    y_att    = (float*)carve((size_t)P * ODIM * 4);
    float*    h_sel    = (float*)carve((size_t)P * HDIM * 4);
    int*      h_ind    = (int*)carve((size_t)P * 4);
    _Float16* W8       = (_Float16*)carve((size_t)8 * HDIM * W8W * 2);
    float*    energyB  = (float*)carve((size_t)P * ESTR * 4);
    float*    x_ext    = (float*)carve((size_t)P * 2 * ODIM * 4);
    float*    lossPart = (float*)carve((size_t)5 * nblkLoss * 4);
    int*      cntPart  = (int*)carve((size_t)ncb * 4);

    const int n8 = 8 * HDIM * W8W;
    const int initThreads = (n8 > P * ODIM) ? n8 : P * ODIM;
    k_init<<<(initThreads + 255) / 256, 256, 0, stream>>>(x, y, W_enc, x_res, y_res, W8, P);
    k_count<<<ncb, 256, 0, stream>>>(y, P * ODIM, cntPart);
    for (int it = 0; it < 5; ++it) {
        k_selector<<<P, 160, 0, stream>>>(x_res, y_res, y_att, P);
        k_energy<<<dim3(nblk16, 8), 256, 0, stream>>>(y_att, b_enc, W8, energyB, P);
        k_argmax_hsel<<<nblk16, 256, 0, stream>>>(y_att, W_enc, b_enc, energyB,
                                                  h_sel, h_ind, P);
        k_dec_gemm<<<dim3(nblk16, 5), 256, 0, stream>>>(h_sel, W_src, b_src, x_ext, P);
        k_loss<<<nblkLoss, 256, 0, stream>>>(x_ext, h_ind, y, y_res,
                                             lossPart, it, P, nblkLoss);
    }
    k_final<<<1, 64, 0, stream>>>(lossPart, cntPart, nblkLoss, ncb, (float*)d_out);
}
